// PerformerLM_GO_29033978921576
// MI455X (gfx1250) — compile-verified, matrix operand order checked by reading, not run on hardware
//
#include <hip/hip_runtime.h>
#include <hip/hip_bf16.h>

// ---------------------------------------------------------------------------
// PerformerLM for MI455X (gfx1250, wave32, WMMA).
// All large GEMMs run through v_wmma_f32_16x16x32_bf16 (bf16 in, f32 acc).
// Every GEMM operand is stored K-contiguous so each 16x32 fragment is two
// aligned global_load_b128 per lane (no gathers, no runtime trans branches).
// LN / exp / stabilizers / segment-sums stay in f32 VALU.
// ---------------------------------------------------------------------------

constexpr int B_     = 2;
constexpr int N_     = 8192;
constexpr int DIM_   = 256;
constexpr int HEADS_ = 8;
constexpr int DH_    = 64;
constexpr int INNER_ = HEADS_ * DH_;   // 512
constexpr int DEPTH_ = 2;
constexpr int NB_    = 256;
constexpr int FF_    = 4 * DIM_;       // 1024
constexpr int E_     = 131072;
constexpr float KEPS_   = 1e-4f;
constexpr float LNEPS_  = 1e-5f;
constexpr float DN_     = 0.35355339059327373f;   // DH^-0.25
constexpr float RATIO_  = 0.0625f;                // NB^-0.5

typedef __attribute__((ext_vector_type(16))) __bf16 v16bf;
typedef __attribute__((ext_vector_type(8)))  __bf16 v8bf;
typedef __attribute__((ext_vector_type(8)))  float  v8f;

__device__ __forceinline__ float gelu_tanh(float x) {
    float x3 = x * x * x;
    return 0.5f * x * (1.0f + tanhf(0.7978845608028654f * (x + 0.044715f * x3)));
}

__device__ __forceinline__ v16bf load_frag(const __bf16* p) {
    // elements 0..7 -> K = base+0..7 ; elements 8..15 -> K = base+16..23
    v8bf lo = *(const v8bf*)(p);
    v8bf hi = *(const v8bf*)(p + 16);
    return __builtin_shufflevector(lo, hi, 0, 1, 2, 3, 4, 5, 6, 7,
                                           8, 9, 10, 11, 12, 13, 14, 15);
}

// ---------------------------------------------------------------------------
// Batched bf16 WMMA GEMM:  C[bz] = act(A[bz] * B[bz] + bias + resid)
// A: M x K row-major (lda, K-contiguous).  B: N x K row-major (ldb,
// K-contiguous, i.e. "pre-transposed").  Block tile 64x64, 8 waves,
// each wave one 16x32 strip (A-fragment reused across 2 WMMAs).
// cbmode: 0 = Cb normal (ldc);  1 = Cb transposed (Cb[cn*ldc+cm]);
//         2 = Cb v-head layout  Cb[((b*H + cn/DH)*DH + cn%DH)*N + n]
// ---------------------------------------------------------------------------
__global__ __launch_bounds__(256)
void plm_gemm(const __bf16* __restrict__ A, const __bf16* __restrict__ Bm,
              const float* __restrict__ bias, const float* __restrict__ resid,
              float* __restrict__ C, __bf16* __restrict__ Cb,
              int M, int Nn, int K,
              int lda, int ldb, int ldc,
              long sA0, long sA1, int hdivA,
              long sB0,
              long sC0, long sC1, int hdivC,
              int act, int cbmode, float bscale)
{
    const int bz    = blockIdx.z;
    const long offA = (long)(bz / hdivA) * sA0 + (long)(bz % hdivA) * sA1;
    const long offB = (long)bz * sB0;
    const long offC = (long)(bz / hdivC) * sC0 + (long)(bz % hdivC) * sC1;

    const int wave = threadIdx.x >> 5;
    const int lane = threadIdx.x & 31;
    const int wm = wave & 3;        // 4 wave-rows  -> 64 rows
    const int wn = wave >> 2;       // 2 wave-cols  -> 2 x 32 cols
    const int m0 = blockIdx.x * 64 + wm * 16;
    const int n0 = blockIdx.y * 64 + wn * 32;
    if (m0 >= M || n0 >= Nn) return;

    const int rc    = lane & 15;        // row (A) / col (B,C) within tile
    const int khalf = (lane >> 4) * 8;  // K sub-half select

    const __bf16* pA  = A  + offA + (long)(m0 + rc) * lda + khalf;
    const __bf16* pB0 = Bm + offB + (long)(n0 + rc) * ldb + khalf;
    const __bf16* pB1 = Bm + offB + (long)(n0 + 16 + rc) * ldb + khalf;

    v8f c0 = {};
    v8f c1 = {};

    for (int k0 = 0; k0 < K; k0 += 32) {
        const v16bf a  = load_frag(pA  + k0);
        const v16bf b0 = load_frag(pB0 + k0);
        const v16bf b1 = load_frag(pB1 + k0);
        c0 = __builtin_amdgcn_wmma_f32_16x16x32_bf16(false, a, false, b0, (short)0, c0, false, false);
        c1 = __builtin_amdgcn_wmma_f32_16x16x32_bf16(false, a, false, b1, (short)0, c1, false, false);
    }

    // C fragment: element r -> row m0 + r + 8*(lane>>4), col n0 + (lane&15)
    const int cn0 = n0 + rc;
#pragma unroll
    for (int r = 0; r < 8; r++) {
        const int cm = m0 + r + 8 * (lane >> 4);
#pragma unroll
        for (int half = 0; half < 2; half++) {
            const int cn = cn0 + half * 16;
            if (cn >= Nn) continue;
            float v = half ? c1[r] : c0[r];
            if (cbmode == 0) {
                const long ci = offC + (long)cm * ldc + cn;
                if (bias)  v += bias[cn];
                if (resid) v += resid[ci];
                if (act == 1) v = gelu_tanh(v);
                if (C)  C[ci]  = v;
                if (Cb) Cb[ci] = (__bf16)(v * bscale);
            } else if (cbmode == 1) {
                Cb[offC + (long)cn * ldc + cm] = (__bf16)(v * bscale);
            } else {
                const long b = cm / N_;
                const long n = cm % N_;
                const long idx = ((b * HEADS_ + cn / DH_) * (long)DH_ + (cn % DH_)) * N_ + n;
                Cb[idx] = (__bf16)(v * bscale);
            }
        }
    }
}

// ------------------------------- small kernels ------------------------------

__global__ void plm_zero(float* p, long n) {
    long i = (long)blockIdx.x * blockDim.x + threadIdx.x;
    for (; i < n; i += (long)gridDim.x * blockDim.x) p[i] = 0.0f;
}

__global__ void plm_cvt(const float* __restrict__ s, __bf16* __restrict__ d, long n) {
    long i = (long)blockIdx.x * blockDim.x + threadIdx.x;
    for (; i < n; i += (long)gridDim.x * blockDim.x) d[i] = (__bf16)s[i];
}

// transpose-convert: src is K x N row-major f32, dst is N x K row-major bf16
__global__ void plm_cvt_t(const float* __restrict__ s, __bf16* __restrict__ d,
                          int K, int N) {
    long i = (long)blockIdx.x * blockDim.x + threadIdx.x;
    const long total = (long)K * N;
    for (; i < total; i += (long)gridDim.x * blockDim.x) {
        const int k = (int)(i / N);
        const int n = (int)(i % N);
        d[(long)n * K + k] = (__bf16)s[i];
    }
}

__global__ void plm_deg(const float* __restrict__ ew, const int* __restrict__ col,
                        float* __restrict__ deg) {
    int e = blockIdx.x * blockDim.x + threadIdx.x;
    if (e < E_) atomicAdd(&deg[col[e]], ew[e]);
}

__global__ void plm_dinv(float* deg) {
    int n = blockIdx.x * blockDim.x + threadIdx.x;
    if (n < N_) {
        float d = deg[n];
        deg[n] = (d > 0.0f) ? rsqrtf(fmaxf(d, 1e-12f)) : 0.0f;
    }
}

__global__ __launch_bounds__(256)
void plm_agg(const float* __restrict__ ew, const int* __restrict__ row,
             const int* __restrict__ col, const float* __restrict__ dinv,
             const float* __restrict__ gemb, float* __restrict__ agg) {
    const int e = blockIdx.x;
    const int d = threadIdx.x;
    const int r = row[e], c = col[e];
    const float en = dinv[r] * ew[e] * dinv[c];
    atomicAdd(&agg[(long)c * DIM_ + d], gemb[(long)r * DIM_ + d] * en);
}

__global__ void plm_embed_h(const float* __restrict__ x, const float* __restrict__ w1,
                            const float* __restrict__ b1, __bf16* __restrict__ hb) {
    long i = (long)blockIdx.x * blockDim.x + threadIdx.x;
    const long total = (long)B_ * N_ * DIM_;
    for (; i < total; i += (long)gridDim.x * blockDim.x) {
        const long t = i / DIM_;
        const int  d = (int)(i % DIM_);
        hb[i] = (__bf16)fmaxf(0.0f, x[t] * w1[d] + b1[d]);
    }
}

__global__ void plm_addposgo(float* __restrict__ z, const float* __restrict__ pos,
                             const float* __restrict__ go) {
    long i = (long)blockIdx.x * blockDim.x + threadIdx.x;
    const long total = (long)B_ * N_ * DIM_;
    const long nd_sz = (long)N_ * DIM_;
    for (; i < total; i += (long)gridDim.x * blockDim.x) {
        const long nd = i % nd_sz;
        z[i] += pos[nd] + go[nd];
    }
}

__global__ __launch_bounds__(256)
void plm_layernorm(const float* __restrict__ x, const float* __restrict__ g,
                   const float* __restrict__ b, float* __restrict__ yf,
                   __bf16* __restrict__ yb) {
    __shared__ float red[256];
    const int row = blockIdx.x;
    const int t   = threadIdx.x;
    const long base = (long)row * DIM_;
    const float v = x[base + t];
    red[t] = v;
    __syncthreads();
    for (int s = 128; s > 0; s >>= 1) { if (t < s) red[t] += red[t + s]; __syncthreads(); }
    const float mean = red[0] / DIM_;
    __syncthreads();
    const float d = v - mean;
    red[t] = d * d;
    __syncthreads();
    for (int s = 128; s > 0; s >>= 1) { if (t < s) red[t] += red[t + s]; __syncthreads(); }
    const float r = rsqrtf(red[0] / DIM_ + LNEPS_);
    const float out = d * r * g[t] + b[t];
    if (yf) yf[base + t] = out;
    if (yb) yb[base + t] = (__bf16)out;
}

// diag = 0.5*sum((q*dn)^2) (input is already dn-scaled bf16); rmax = row max of dd
__global__ void plm_rowstat(const __bf16* __restrict__ qb, const float* __restrict__ dd,
                            float* __restrict__ diag, float* __restrict__ rmax) {
    long r = (long)blockIdx.x * blockDim.x + threadIdx.x;
    const long total = (long)B_ * HEADS_ * N_;
    if (r >= total) return;
    const int n = (int)(r % N_);
    const int h = (int)((r / N_) % HEADS_);
    const int b = (int)(r / ((long)N_ * HEADS_));
    const long qoff = ((long)b * N_ + n) * INNER_ + (long)h * DH_;
    float s = 0.0f;
    for (int d = 0; d < DH_; d++) { const float q = (float)qb[qoff + d]; s += q * q; }
    diag[r] = 0.5f * s;
    float m = -1e30f;
    const long doff = r * NB_;
    for (int j = 0; j < NB_; j++) m = fmaxf(m, dd[doff + j]);
    rmax[r] = m;
}

__global__ __launch_bounds__(256)
void plm_colmax(const float* __restrict__ rmax, float* __restrict__ stab) {
    __shared__ float red[256];
    const int bh = blockIdx.x;
    float m = -1e30f;
    for (int n = threadIdx.x; n < N_; n += 256) m = fmaxf(m, rmax[(long)bh * N_ + n]);
    red[threadIdx.x] = m;
    __syncthreads();
    for (int s = 128; s > 0; s >>= 1) {
        if (threadIdx.x < s) red[threadIdx.x] = fmaxf(red[threadIdx.x], red[threadIdx.x + s]);
        __syncthreads();
    }
    if (threadIdx.x == 0) stab[bh] = red[0];
}

// qf/kf = ratio*(exp(dd - diag - stab) + eps).
// transOut=0: out[(bh,n,m)] (qf);  transOut=1: out[(bh,m,n)] (kf, K-contiguous)
__global__ void plm_apply(const float* __restrict__ dd, const float* __restrict__ diag,
                          const float* __restrict__ rmax, const float* __restrict__ stabk,
                          __bf16* __restrict__ out, int useCol, int transOut) {
    long i = (long)blockIdx.x * blockDim.x + threadIdx.x;
    const long total = (long)B_ * HEADS_ * N_ * NB_;
    for (; i < total; i += (long)gridDim.x * blockDim.x) {
        const long r = i / NB_;
        const long m = i % NB_;
        const float stab = useCol ? stabk[r / N_] : rmax[r];
        const float v = RATIO_ * (expf(dd[i] - diag[r] - stab) + KEPS_);
        if (transOut) {
            const long bh = r / N_;
            const long n  = r % N_;
            out[((long)bh * NB_ + m) * N_ + n] = (__bf16)v;
        } else {
            out[i] = (__bf16)v;
        }
    }
}

// ksum over n of kf_t (b,h,m,n): contiguous reduction
__global__ void plm_ksum(const __bf16* __restrict__ kft, float* __restrict__ ksum) {
    const int i = blockIdx.x * blockDim.x + threadIdx.x;
    if (i >= B_ * HEADS_ * NB_) return;
    const long base = (long)i * N_;
    float s = 0.0f;
    for (int n = 0; n < N_; n++) s += (float)kft[base + n];
    ksum[i] = s;
}

__global__ void plm_denom(const __bf16* __restrict__ qf, const float* __restrict__ ksum,
                          float* __restrict__ denom) {
    long r = (long)blockIdx.x * blockDim.x + threadIdx.x;
    const long total = (long)B_ * HEADS_ * N_;
    if (r >= total) return;
    const long bh = r / N_;
    const long base = r * NB_;
    float s = 0.0f;
    for (int m = 0; m < NB_; m++) s += (float)qf[base + m] * ksum[bh * NB_ + m];
    denom[r] = s;
}

__global__ void plm_divide(const float* __restrict__ o, const float* __restrict__ denom,
                           __bf16* __restrict__ ob) {
    long i = (long)blockIdx.x * blockDim.x + threadIdx.x;
    const long total = (long)B_ * N_ * INNER_;
    for (; i < total; i += (long)gridDim.x * blockDim.x) {
        const long b   = i / ((long)N_ * INNER_);
        const long rem = i % ((long)N_ * INNER_);
        const long n   = rem / INNER_;
        const long h   = (rem % INNER_) / DH_;
        const long r   = (b * HEADS_ + h) * N_ + n;
        ob[i] = (__bf16)(o[i] / denom[r]);
    }
}

__global__ void plm_out(const float* __restrict__ yf, const float* __restrict__ ow,
                        const float* __restrict__ ob, float* __restrict__ out) {
    const long t = (long)blockIdx.x * blockDim.x + threadIdx.x;
    if (t >= (long)B_ * N_) return;
    float s = 0.0f;
    const long base = t * DIM_;
    for (int d = 0; d < DIM_; d++) s += yf[base + d] * ow[d];
    out[t] = s + ob[0];
}

// ---------------------------------------------------------------------------

extern "C" void kernel_launch(void* const* d_in, const int* in_sizes, int n_in,
                              void* d_out, int out_size, void* d_ws, size_t ws_size,
                              hipStream_t stream) {
    (void)in_sizes; (void)n_in; (void)out_size; (void)ws_size;
    const float* x        = (const float*)d_in[0];
    const float* mlp_w1   = (const float*)d_in[1];
    const float* mlp_b1   = (const float*)d_in[2];
    const float* mlp_w2   = (const float*)d_in[3];
    const float* mlp_b2   = (const float*)d_in[4];
    const float* pos_tab  = (const float*)d_in[5];
    const float* gnn_emb  = (const float*)d_in[6];
    const float* gnn_w    = (const float*)d_in[7];
    const float* gnn_b    = (const float*)d_in[8];
    const float* edge_w   = (const float*)d_in[9];
    const float* proj     = (const float*)d_in[10];
    const float* ln1_g    = (const float*)d_in[11];
    const float* ln1_b    = (const float*)d_in[12];
    const float* wq       = (const float*)d_in[13];
    const float* wk       = (const float*)d_in[14];
    const float* wv       = (const float*)d_in[15];
    const float* wo       = (const float*)d_in[16];
    const float* bo       = (const float*)d_in[17];
    const float* ln2_g    = (const float*)d_in[18];
    const float* ln2_b    = (const float*)d_in[19];
    const float* ffw1     = (const float*)d_in[20];
    const float* ffb1     = (const float*)d_in[21];
    const float* ffw2     = (const float*)d_in[22];
    const float* ffb2     = (const float*)d_in[23];
    const float* normf_g  = (const float*)d_in[24];
    const float* normf_b  = (const float*)d_in[25];
    const float* out_w    = (const float*)d_in[26];
    const float* out_b    = (const float*)d_in[27];
    const int*   eidx     = (const int*)d_in[28];
    float* out = (float*)d_out;

    // ---- workspace bump allocator (needs ~480 MB) ----
    char* wsp = (char*)d_ws;
    auto alloc = [&](size_t bytes) -> void* {
        void* p = (void*)wsp;
        wsp += (bytes + 255) & ~(size_t)255;
        return p;
    };
    const long BN  = (long)B_ * N_;
    const long BHN = (long)B_ * HEADS_ * N_;

    // all weight copies are stored TRANSPOSED (N_out x K, K-contiguous)
    __bf16* wq_t   = (__bf16*)alloc((size_t)DEPTH_ * DIM_ * INNER_ * 2);
    __bf16* wk_t   = (__bf16*)alloc((size_t)DEPTH_ * DIM_ * INNER_ * 2);
    __bf16* wv_t   = (__bf16*)alloc((size_t)DEPTH_ * DIM_ * INNER_ * 2);
    __bf16* wo_t   = (__bf16*)alloc((size_t)DEPTH_ * INNER_ * DIM_ * 2);
    __bf16* ffw1_t = (__bf16*)alloc((size_t)DEPTH_ * DIM_ * FF_ * 2);
    __bf16* ffw2_t = (__bf16*)alloc((size_t)DEPTH_ * FF_ * DIM_ * 2);
    __bf16* mw2_t  = (__bf16*)alloc((size_t)DIM_ * DIM_ * 2);
    __bf16* gw_t   = (__bf16*)alloc((size_t)DIM_ * DIM_ * 2);
    __bf16* proj_b = (__bf16*)alloc((size_t)NB_ * DH_ * 2);   // already N x K
    float*  deg    = (float*)alloc((size_t)N_ * 4);           // becomes dinv
    float*  agg    = (float*)alloc((size_t)N_ * DIM_ * 4);
    __bf16* agg_b  = (__bf16*)alloc((size_t)N_ * DIM_ * 2);
    float*  go     = (float*)alloc((size_t)N_ * DIM_ * 4);
    __bf16* h_b    = (__bf16*)alloc((size_t)BN * DIM_ * 2);
    float*  z      = (float*)alloc((size_t)BN * DIM_ * 4);
    __bf16* y_b    = (__bf16*)alloc((size_t)BN * DIM_ * 2);
    float*  yf     = (float*)alloc((size_t)BN * DIM_ * 4);
    __bf16* qb     = (__bf16*)alloc((size_t)BN * INNER_ * 2); // (b,n,h,dh)
    __bf16* kb     = (__bf16*)alloc((size_t)BN * INNER_ * 2); // (b,n,h,dh)
    __bf16* vt     = (__bf16*)alloc((size_t)BN * INNER_ * 2); // (b,h,dh,n)
    float*  dd     = (float*)alloc((size_t)BHN * NB_ * 4);    // reused for q then k
    __bf16* qf_b   = (__bf16*)alloc((size_t)BHN * NB_ * 2);   // (b,h,n,m)
    __bf16* kf_t   = (__bf16*)alloc((size_t)BHN * NB_ * 2);   // (b,h,m,n)
    float*  diag   = (float*)alloc((size_t)BHN * 4);
    float*  rmax   = (float*)alloc((size_t)BHN * 4);
    float*  stabk  = (float*)alloc((size_t)B_ * HEADS_ * 4);
    float*  ksum   = (float*)alloc((size_t)B_ * HEADS_ * NB_ * 4);
    float*  denom  = (float*)alloc((size_t)BHN * 4);
    __bf16* ctx_t  = (__bf16*)alloc((size_t)B_ * HEADS_ * NB_ * DH_ * 2); // (b,h,d,m)
    float*  o      = (float*)alloc((size_t)BN * INNER_ * 4);  // (b,n,h,dh)
    __bf16* o_b    = (__bf16*)alloc((size_t)BN * INNER_ * 2);
    __bf16* ffh_b  = (__bf16*)alloc((size_t)BN * FF_ * 2);

    const int TPB = 256;
    auto g1 = [&](long n) { return dim3((unsigned)((n + TPB - 1) / TPB)); };

    // ---- weight conversions (f32 -> bf16, transposed to N_out x K) ----
    for (int l = 0; l < DEPTH_; l++) {
        plm_cvt_t<<<g1((long)DIM_*INNER_), TPB, 0, stream>>>(wq + (long)l*DIM_*INNER_,
                                                             wq_t + (long)l*INNER_*DIM_, DIM_, INNER_);
        plm_cvt_t<<<g1((long)DIM_*INNER_), TPB, 0, stream>>>(wk + (long)l*DIM_*INNER_,
                                                             wk_t + (long)l*INNER_*DIM_, DIM_, INNER_);
        plm_cvt_t<<<g1((long)DIM_*INNER_), TPB, 0, stream>>>(wv + (long)l*DIM_*INNER_,
                                                             wv_t + (long)l*INNER_*DIM_, DIM_, INNER_);
        plm_cvt_t<<<g1((long)INNER_*DIM_), TPB, 0, stream>>>(wo + (long)l*INNER_*DIM_,
                                                             wo_t + (long)l*DIM_*INNER_, INNER_, DIM_);
        plm_cvt_t<<<g1((long)DIM_*FF_),    TPB, 0, stream>>>(ffw1 + (long)l*DIM_*FF_,
                                                             ffw1_t + (long)l*FF_*DIM_, DIM_, FF_);
        plm_cvt_t<<<g1((long)FF_*DIM_),    TPB, 0, stream>>>(ffw2 + (long)l*FF_*DIM_,
                                                             ffw2_t + (long)l*DIM_*FF_, FF_, DIM_);
    }
    plm_cvt_t<<<g1((long)DIM_*DIM_), TPB, 0, stream>>>(mlp_w2, mw2_t, DIM_, DIM_);
    plm_cvt_t<<<g1((long)DIM_*DIM_), TPB, 0, stream>>>(gnn_w,  gw_t,  DIM_, DIM_);
    plm_cvt  <<<g1((long)NB_*DH_),   TPB, 0, stream>>>(proj, proj_b, (long)NB_*DH_);

    // ---- GNN aggregation ----
    plm_zero<<<g1(N_), TPB, 0, stream>>>(deg, N_);
    plm_zero<<<g1((long)N_*DIM_), TPB, 0, stream>>>(agg, (long)N_*DIM_);
    plm_deg<<<g1(E_), TPB, 0, stream>>>(edge_w, eidx + E_, deg);
    plm_dinv<<<g1(N_), TPB, 0, stream>>>(deg);
    plm_agg<<<E_, DIM_, 0, stream>>>(edge_w, eidx, eidx + E_, deg, gnn_emb, agg);
    plm_cvt<<<g1((long)N_*DIM_), TPB, 0, stream>>>(agg, agg_b, (long)N_*DIM_);
    // go = agg @ gnn_w + gnn_b
    plm_gemm<<<dim3(N_/64, DIM_/64, 1), TPB, 0, stream>>>(
        agg_b, gw_t, gnn_b, nullptr, go, nullptr,
        N_, DIM_, DIM_, DIM_, DIM_, DIM_,
        0, 0, 1, 0, 0, 0, 1, 0, 0, 1.0f);

    // ---- token embedding ----
    plm_embed_h<<<g1(BN*DIM_), TPB, 0, stream>>>(x, mlp_w1, mlp_b1, h_b);
    plm_gemm<<<dim3(BN/64, DIM_/64, 1), TPB, 0, stream>>>(
        h_b, mw2_t, mlp_b2, nullptr, z, nullptr,
        (int)BN, DIM_, DIM_, DIM_, DIM_, DIM_,
        0, 0, 1, 0, 0, 0, 1, 0, 0, 1.0f);
    plm_addposgo<<<g1(BN*DIM_), TPB, 0, stream>>>(z, pos_tab, go);

    const int BH = B_ * HEADS_;
    for (int l = 0; l < DEPTH_; l++) {
        // --- attention ---
        plm_layernorm<<<(unsigned)BN, TPB, 0, stream>>>(z, ln1_g + l*DIM_, ln1_b + l*DIM_,
                                                        nullptr, y_b);
        // q/k/v projections; q,k pre-scaled by DH^-0.25; v stored (b,h,d,n)
        plm_gemm<<<dim3(BN/64, INNER_/64, 1), TPB, 0, stream>>>(
            y_b, wq_t + (long)l*INNER_*DIM_, nullptr, nullptr, nullptr, qb,
            (int)BN, INNER_, DIM_, DIM_, DIM_, INNER_,
            0, 0, 1, 0, 0, 0, 1, 0, 0, DN_);
        plm_gemm<<<dim3(BN/64, INNER_/64, 1), TPB, 0, stream>>>(
            y_b, wk_t + (long)l*INNER_*DIM_, nullptr, nullptr, nullptr, kb,
            (int)BN, INNER_, DIM_, DIM_, DIM_, INNER_,
            0, 0, 1, 0, 0, 0, 1, 0, 0, DN_);
        plm_gemm<<<dim3(BN/64, INNER_/64, 1), TPB, 0, stream>>>(
            y_b, wv_t + (long)l*INNER_*DIM_, nullptr, nullptr, nullptr, vt,
            (int)BN, INNER_, DIM_, DIM_, DIM_, N_,
            0, 0, 1, 0, 0, 0, 1, 0, 2 /*v-head layout*/, 1.0f);

        // q features: dd = (q*dn) @ proj^T, per (b,h)
        plm_gemm<<<dim3(N_/64, NB_/64, BH), TPB, 0, stream>>>(
            qb, proj_b, nullptr, nullptr, dd, nullptr,
            N_, NB_, DH_, INNER_, DH_, NB_,
            (long)N_*INNER_, DH_, HEADS_, 0, (long)N_*NB_, 0, 1, 0, 0, 1.0f);
        plm_rowstat<<<g1(BHN), TPB, 0, stream>>>(qb, dd, diag, rmax);
        plm_apply<<<g1(BHN*NB_), TPB, 0, stream>>>(dd, diag, rmax, stabk, qf_b, 0, 0);

        // k features (dd buffer reused); kf stored transposed (b,h,m,n)
        plm_gemm<<<dim3(N_/64, NB_/64, BH), TPB, 0, stream>>>(
            kb, proj_b, nullptr, nullptr, dd, nullptr,
            N_, NB_, DH_, INNER_, DH_, NB_,
            (long)N_*INNER_, DH_, HEADS_, 0, (long)N_*NB_, 0, 1, 0, 0, 1.0f);
        plm_rowstat<<<g1(BHN), TPB, 0, stream>>>(kb, dd, diag, rmax);
        plm_colmax<<<BH, TPB, 0, stream>>>(rmax, stabk);
        plm_apply<<<g1(BHN*NB_), TPB, 0, stream>>>(dd, diag, rmax, stabk, kf_t, 1, 1);

        plm_ksum<<<g1(BH*NB_), TPB, 0, stream>>>(kf_t, ksum);
        plm_denom<<<g1(BHN), TPB, 0, stream>>>(qf_b, ksum, denom);

        // ctx[b,h] = kf^T @ v : A = kf_t (NB x N), B = vt (DH x N) -> ctx_t (DH x NB)
        plm_gemm<<<dim3(NB_/64, 1, BH), TPB, 0, stream>>>(
            kf_t, vt, nullptr, nullptr, nullptr, ctx_t,
            NB_, DH_, N_, N_, N_, NB_,
            (long)NB_*N_, 0, 1, (long)DH_*N_, (long)DH_*NB_, 0, 1, 0, 1 /*transposed*/, 1.0f);
        // o[b,h] = qf @ ctx : A = qf (N x NB), B = ctx_t (DH x NB) -> o (b,n,h,dh)
        plm_gemm<<<dim3(N_/64, 1, BH), TPB, 0, stream>>>(
            qf_b, ctx_t, nullptr, nullptr, o, nullptr,
            N_, DH_, NB_, NB_, NB_, INNER_,
            (long)N_*NB_, 0, 1, (long)DH_*NB_, (long)N_*INNER_, DH_, HEADS_, 0, 0, 1.0f);
        plm_divide<<<g1(BN*INNER_), TPB, 0, stream>>>(o, denom, o_b);

        // z += o @ wo + bo
        plm_gemm<<<dim3(BN/64, DIM_/64, 1), TPB, 0, stream>>>(
            o_b, wo_t + (long)l*DIM_*INNER_, bo + l*DIM_, z, z, nullptr,
            (int)BN, DIM_, INNER_, INNER_, INNER_, DIM_,
            0, 0, 1, 0, 0, 0, 1, 0, 0, 1.0f);

        // --- feed-forward ---
        plm_layernorm<<<(unsigned)BN, TPB, 0, stream>>>(z, ln2_g + l*DIM_, ln2_b + l*DIM_,
                                                        nullptr, y_b);
        plm_gemm<<<dim3(BN/64, FF_/64, 1), TPB, 0, stream>>>(
            y_b, ffw1_t + (long)l*FF_*DIM_, ffb1 + l*FF_, nullptr, nullptr, ffh_b,
            (int)BN, FF_, DIM_, DIM_, DIM_, FF_,
            0, 0, 1, 0, 0, 0, 1, 1 /*gelu*/, 0, 1.0f);
        plm_gemm<<<dim3(BN/64, DIM_/64, 1), TPB, 0, stream>>>(
            ffh_b, ffw2_t + (long)l*DIM_*FF_, ffb2 + l*DIM_, z, z, nullptr,
            (int)BN, DIM_, FF_, FF_, FF_, DIM_,
            0, 0, 1, 0, 0, 0, 1, 0, 0, 1.0f);
    }

    // ---- final LN + projection to scalar ----
    plm_layernorm<<<(unsigned)BN, TPB, 0, stream>>>(z, normf_g, normf_b, yf, nullptr);
    plm_out<<<g1(BN), TPB, 0, stream>>>(yf, out_w, out_b, out);
}
